// NlotmImageAutoEncoder_15178414424134
// MI455X (gfx1250) — matrix operand order repaired and checked by reading, not compiled, not run on hardware
//
#include <hip/hip_runtime.h>

typedef __attribute__((ext_vector_type(16))) _Float16 v16h;
typedef __attribute__((ext_vector_type(8)))  _Float16 v8h;
typedef __attribute__((ext_vector_type(4)))  _Float16 v4h;
typedef __attribute__((ext_vector_type(8)))  float    v8f;

#define D_MODEL     2048
#define D_BLOCK     256
#define NUM_BLOCKS  8
#define NUM_PROT    1024
#define NTOK        16384            // B*N = 64*256
#define ROWS_PER_WG 128              // 8 waves x 16 rows
#define PROTO_STAGE 64
#define LDS_STRIDE  264              // halfs: 256 + 8 pad -> conflict-free b128 reads

#define EMB_ELEMS   (NTOK * D_MODEL)                 // 33,554,432
#define IDX_OUT_OFF EMB_ELEMS
#define VQ_OFF      (EMB_ELEMS + NTOK * NUM_BLOCKS)  // 33,685,504
#define COMMIT_OFF  (VQ_OFF + 1)

// workspace layout (float units)
#define WS_K2_OFF   0                                // 8192 floats
#define WS_IDX_OFF  8192                             // 131072 ints
#define WS_PART_OFF (8192 + NTOK * NUM_BLOCKS)       // 32768 floats
#define WS_MEMH_OFF (WS_PART_OFF + 32768)            // 2M halfs (4MB), 16B aligned
#define NUM_GATHER_BLOCKS (NTOK * NUM_BLOCKS / 4)    // 32768

// -------- phase 0: k2[p] = ||mem[p]||^2 and mem fp32 -> f16, one wave/proto --
__global__ __launch_bounds__(256) void prep_kernel(const float* __restrict__ mem,
                                                   float* __restrict__ k2,
                                                   _Float16* __restrict__ memh) {
  const int proto = blockIdx.x * 8 + (threadIdx.x >> 5);
  const int lane  = threadIdx.x & 31;
  const float4* r = (const float4*)(mem + (size_t)proto * D_BLOCK);
  v4h* w = (v4h*)(memh + (size_t)proto * D_BLOCK);
  float s = 0.f;
  #pragma unroll
  for (int i = 0; i < 2; ++i) {
    const float4 v = r[lane + 32 * i];
    s = __builtin_fmaf(v.x, v.x, s);
    s = __builtin_fmaf(v.y, v.y, s);
    s = __builtin_fmaf(v.z, v.z, s);
    s = __builtin_fmaf(v.w, v.w, s);
    v4h h;
    h.x = (_Float16)v.x; h.y = (_Float16)v.y;
    h.z = (_Float16)v.z; h.w = (_Float16)v.w;
    w[lane + 32 * i] = h;
  }
  #pragma unroll
  for (int mask = 16; mask >= 1; mask >>= 1) s += __shfl_xor(s, mask, 32);
  if (lane == 0) k2[proto] = s;
}

// issue the 8 per-thread async b128 loads for one 64-proto stage
__device__ __forceinline__ void stage_async(uint32_t ldsBufBase,
                                            unsigned long long sbase,
                                            int pbase, int tid) {
  #pragma unroll
  for (int it = 0; it < 8; ++it) {
    const int f   = it * 256 + tid;        // b128 chunk index 0..2047
    const int pr  = f >> 5;                // proto within stage (32 chunks/row)
    const int c16 = f & 31;
    const uint32_t ldsAddr = ldsBufBase + (uint32_t)(pr * (LDS_STRIDE * 2) + c16 * 16);
    const uint32_t goff    = (uint32_t)((pbase + pr) * (D_BLOCK * 2) + c16 * 16);
    asm volatile("global_load_async_to_lds_b128 %0, %1, %2"
                 :: "v"(ldsAddr), "v"(goff), "s"(sbase) : "memory");
  }
}

// -------- phase 1: argmin_p (k2[p] - 2*q.k), f16 WMMA, double-buffered async -
__global__ __launch_bounds__(256) void vq_argmin_kernel(
    const float* __restrict__ q, const _Float16* __restrict__ memh,
    const float* __restrict__ k2, int* __restrict__ idxOut) {
  __shared__ _Float16 ldsB[2][PROTO_STAGE * LDS_STRIDE];
  __shared__ float    lds_k2[2][PROTO_STAGE];

  const int tid  = threadIdx.x;
  const int lane = tid & 31;
  const int wave = tid >> 5;           // 0..7
  const int half = lane >> 4;          // 0 or 1
  const int l16  = lane & 15;
  const int m    = blockIdx.y;
  const int rowbase = blockIdx.x * ROWS_PER_WG + wave * 16;

  const uint32_t ldsBase = (uint32_t)(uintptr_t)(&ldsB[0][0]);
  const unsigned long long sbase =
      (unsigned long long)(uintptr_t)(memh + (size_t)m * NUM_PROT * D_BLOCK);
  const float* k2m = k2 + m * NUM_PROT;

  // kick off stage 0 before doing anything else
  stage_async(ldsBase, sbase, 0, tid);
  if (tid < PROTO_STAGE) lds_k2[0][tid] = k2m[tid];

  // ---- A fragments: 16 q-rows x K=256, f32 -> f16, kept in registers -------
  // (overlaps with the in-flight async stage)
  // 16-bit A 16x32 layout: elems 0..7 -> K = half*8 + 0..7 ; elems 8..15 -> K = 16 + half*8 + 0..7
  const float* qrow = q + (size_t)(rowbase + l16) * D_MODEL + m * D_BLOCK;
  v16h a[8];
  #pragma unroll
  for (int kc = 0; kc < 8; ++kc) {
    const int d0 = kc * 32 + half * 8;
    const float4 f0 = *(const float4*)(qrow + d0);
    const float4 f1 = *(const float4*)(qrow + d0 + 4);
    const float4 f2 = *(const float4*)(qrow + d0 + 16);
    const float4 f3 = *(const float4*)(qrow + d0 + 20);
    a[kc][0]=(_Float16)f0.x;  a[kc][1]=(_Float16)f0.y;  a[kc][2]=(_Float16)f0.z;  a[kc][3]=(_Float16)f0.w;
    a[kc][4]=(_Float16)f1.x;  a[kc][5]=(_Float16)f1.y;  a[kc][6]=(_Float16)f1.z;  a[kc][7]=(_Float16)f1.w;
    a[kc][8]=(_Float16)f2.x;  a[kc][9]=(_Float16)f2.y;  a[kc][10]=(_Float16)f2.z; a[kc][11]=(_Float16)f2.w;
    a[kc][12]=(_Float16)f3.x; a[kc][13]=(_Float16)f3.y; a[kc][14]=(_Float16)f3.z; a[kc][15]=(_Float16)f3.w;
  }

  float best[8];
  int   bidx[8];
  #pragma unroll
  for (int r = 0; r < 8; ++r) { best[r] = 3.4e38f; bidx[r] = 0; }

  asm volatile("s_wait_asynccnt 0x0" ::: "memory");
  __syncthreads();   // stage 0 visible to all waves

  for (int pbase = 0; pbase < NUM_PROT; pbase += PROTO_STAGE) {
    const int buf = (pbase >> 6) & 1;
    // issue next stage into the other buffer while we compute on this one;
    // safe: all waves finished reading buf^1 at the barrier ending last iter.
    if (pbase + PROTO_STAGE < NUM_PROT) {
      stage_async(ldsBase + (buf ^ 1) * (uint32_t)(PROTO_STAGE * LDS_STRIDE * 2),
                  sbase, pbase + PROTO_STAGE, tid);
      if (tid < PROTO_STAGE) lds_k2[buf ^ 1][tid] = k2m[pbase + PROTO_STAGE + tid];
    }

    // ---- 4 sub-tiles of 16 prototypes ------------------------------------
    #pragma unroll
    for (int sub = 0; sub < 4; ++sub) {
      // 16-bit B 32x16 layout: lane = column N (= proto), elems 0..7 -> K = half*8 + 0..7 ; elems 8..15 -> K = 16 + half*8 + 0..7
      const _Float16* bp = &ldsB[buf][(sub * 16 + l16) * LDS_STRIDE + half * 8];
      v8f acc = {0.f, 0.f, 0.f, 0.f, 0.f, 0.f, 0.f, 0.f};
      // groups of 4 K-chunks: 8 ds_load_b128 in flight, then 4 WMMAs.
      // loads cannot sink below the WMMAs (all four fragments are consumed),
      // so LDS latency overlaps inside each group.
      #pragma unroll
      for (int g = 0; g < 2; ++g) {
        union Bu { v16h v; v8h h[2]; } bf[4];
        #pragma unroll
        for (int j = 0; j < 4; ++j) {
          const int kc = g * 4 + j;
          bf[j].h[0] = *(const v8h*)(bp + kc * 32);
          bf[j].h[1] = *(const v8h*)(bp + kc * 32 + 16);
        }
        #pragma unroll
        for (int j = 0; j < 4; ++j) {
          acc = __builtin_amdgcn_wmma_f32_16x16x32_f16(
              false, a[g * 4 + j], false, bf[j].v, (short)0, acc, false, false);
        }
      }
      const float kk = lds_k2[buf][sub * 16 + l16];
      const int   pg = pbase + sub * 16 + l16;
      #pragma unroll
      for (int r = 0; r < 8; ++r) {
        const float d = __builtin_fmaf(-2.0f, acc[r], kk);  // q2 const in p: drop it
        if (d < best[r]) { best[r] = d; bidx[r] = pg; }
      }
    }

    // own async loads (and lds_k2 store) done, then all waves rendezvous:
    // next iter's buffer is ready and this buffer is free to overwrite.
    asm volatile("s_wait_asynccnt 0x0" ::: "memory");
    __syncthreads();
  }

  // ---- argmin across the 16 lanes of each half-wave (ties -> lowest idx) ---
  #pragma unroll
  for (int mask = 1; mask < 16; mask <<= 1) {
    #pragma unroll
    for (int r = 0; r < 8; ++r) {
      const float ov = __shfl_xor(best[r], mask, 32);
      const int   oi = __shfl_xor(bidx[r], mask, 32);
      if (ov < best[r] || (ov == best[r] && oi < bidx[r])) { best[r] = ov; bidx[r] = oi; }
    }
  }
  if (l16 == 0) {   // lane 0 -> rows M=0..7, lane 16 -> rows M=8..15
    const int r0 = rowbase + half * 8;
    #pragma unroll
    for (int r = 0; r < 8; ++r)
      idxOut[(size_t)(r0 + r) * NUM_BLOCKS + m] = bidx[r];
  }
}

// -------- phase 2: gather winners (exact fp32), indices, partial commit -----
__global__ __launch_bounds__(256) void gather_kernel(
    const float* __restrict__ q, const float* __restrict__ mem,
    const int* __restrict__ idxWs, float* __restrict__ out,
    float* __restrict__ partials) {
  const int tid  = threadIdx.x;
  const int pair = blockIdx.x * 4 + (tid >> 6);   // flat (n*8 + m)
  const int j    = tid & 63;                      // float4 within 256-d block
  const int n = pair >> 3;
  const int m = pair & 7;
  const int idx = idxWs[pair];
  const float4* src = (const float4*)(mem + ((size_t)m * NUM_PROT + idx) * D_BLOCK);
  const float4* qv  = (const float4*)(q + (size_t)n * D_MODEL + m * D_BLOCK);
  float4*       dst = (float4*)(out + (size_t)n * D_MODEL + m * D_BLOCK);
  const float4 e  = src[j];
  const float4 qq = qv[j];
  dst[j] = e;
  const float dx = e.x - qq.x, dy = e.y - qq.y, dz = e.z - qq.z, dw = e.w - qq.w;
  float s = dx * dx + dy * dy + dz * dz + dw * dw;
  if (j == 0) out[IDX_OUT_OFF + pair] = (float)(idx + (m << 10));

  __shared__ float red[256];
  red[tid] = s;
  __syncthreads();
  #pragma unroll
  for (int st = 128; st > 0; st >>= 1) {
    if (tid < st) red[tid] += red[tid + st];
    __syncthreads();
  }
  if (tid == 0) partials[blockIdx.x] = red[0];
}

// -------- phase 3: deterministic final reduction + scalar losses ------------
__global__ __launch_bounds__(256) void finalize_kernel(
    const float* __restrict__ partials, float* __restrict__ out) {
  __shared__ float red[256];
  float s = 0.f;
  for (int i = threadIdx.x; i < NUM_GATHER_BLOCKS; i += 256) s += partials[i];
  red[threadIdx.x] = s;
  __syncthreads();
  #pragma unroll
  for (int st = 128; st > 0; st >>= 1) {
    if (threadIdx.x < st) red[threadIdx.x] += red[threadIdx.x + st];
    __syncthreads();
  }
  if (threadIdx.x == 0) {
    out[VQ_OFF]     = 0.0f;
    out[COMMIT_OFF] = red[0] / (float)EMB_ELEMS;
  }
}

extern "C" void kernel_launch(void* const* d_in, const int* in_sizes, int n_in,
                              void* d_out, int out_size, void* d_ws, size_t ws_size,
                              hipStream_t stream) {
  const float* q   = (const float*)d_in[0];   // (64,256,2048) f32
  const float* mem = (const float*)d_in[1];   // (8192,256) f32
  float* out = (float*)d_out;
  float* wsf = (float*)d_ws;

  float*     k2       = wsf + WS_K2_OFF;
  int*       idxWs    = (int*)(wsf + WS_IDX_OFF);
  float*     partials = wsf + WS_PART_OFF;
  _Float16*  memh     = (_Float16*)(wsf + WS_MEMH_OFF);

  prep_kernel<<<dim3(NUM_BLOCKS * NUM_PROT / 8), 256, 0, stream>>>(mem, k2, memh);
  vq_argmin_kernel<<<dim3(NTOK / ROWS_PER_WG, NUM_BLOCKS), 256, 0, stream>>>(q, memh, k2, idxWs);
  gather_kernel<<<dim3(NUM_GATHER_BLOCKS), 256, 0, stream>>>(q, mem, idxWs, out, partials);
  finalize_kernel<<<1, 256, 0, stream>>>(partials, out);
}
